// PointerGNN_39195871543578
// MI455X (gfx1250) — matrix operand order repaired and checked by reading, not compiled, not run on hardware
//
#include <hip/hip_runtime.h>
#include <math.h>

// ---------------- problem constants ----------------
#define B_      8
#define N_      4096
#define D_      256
#define HID_    1024
#define L_      6
#define MTOT    (B_ * N_)      // 32768 rows
#define CHUNKM  8192           // row chunk for big intermediates
#define NCHUNK  (MTOT / CHUNKM)

// Use CDNA5 async global->LDS copies (ASYNCcnt) for GEMM A-panel staging.
#define USE_ASYNC_LDS 1

// ---------------- CDNA5 WMMA types ----------------
typedef __attribute__((ext_vector_type(16))) __bf16 v16bf;
typedef __attribute__((ext_vector_type(8)))  float  v8f;

struct U32x8 { uint4 a, b; };

__device__ __forceinline__ unsigned short f32_to_bf16(float f) {
    unsigned int u = __builtin_bit_cast(unsigned int, f);
    u += 0x7FFFu + ((u >> 16) & 1u);     // round-to-nearest-even
    return (unsigned short)(u >> 16);
}

__device__ __forceinline__ float gelu_exact(float x) {
    return 0.5f * x * (1.0f + erff(x * 0.70710678118654752440f));
}

__device__ __forceinline__ v16bf load_frag16(const unsigned short* p0,
                                             const unsigned short* p1) {
    U32x8 t;
    t.a = *(const uint4*)p0;
    t.b = *(const uint4*)p1;
    return __builtin_bit_cast(v16bf, t);
}

#if USE_ASYNC_LDS
// GLOBAL_LOAD_ASYNC_TO_LDS_B128: LDS[vdst_lane] = MEM[vaddr_lane .. +15]
// (ISA 15.18.3 opcode 98; tracked with ASYNCcnt). LDS byte offset is the low
// 32 bits of the generic shared pointer (flat->LDS truncation, ISA 10.2).
__device__ __forceinline__ void async_ld_b128(unsigned int lds_off,
                                              const unsigned short* g) {
    asm volatile("global_load_async_to_lds_b128 %0, %1, off"
                 :: "v"(lds_off), "v"(g)
                 : "memory");
}
__device__ __forceinline__ void async_wait_all() {
    asm volatile("s_wait_asynccnt 0x0" ::: "memory");
}
#endif

// ---------------- weight repack: W[NOUT][K] f32 -> B-fragment-major bf16 ----
// Fragment (nt, kt) covers B[k = kt*32 .. +31][n = nt*16 .. +15] with
// B[k][n] = W[n][k]. Contiguous k (not ISA table order): lanes 0-15 hold K 0..15 of
// column n=lane, lanes 16-31 hold K 16..31 of column n=lane-16.
__global__ void repack_weight(const float* __restrict__ W,
                              unsigned short* __restrict__ out,
                              int NOUT, int K) {
    int t = blockIdx.x * blockDim.x + threadIdx.x;
    int total = (NOUT / 16) * (K / 32) * 32;
    if (t >= total) return;
    int lane = t & 31;
    int frag = t >> 5;
    int ktiles = K >> 5;
    int nt = frag / ktiles, kt = frag % ktiles;
    int n  = nt * 16 + (lane & 15);
    int kb = kt * 32 + ((lane >> 4) << 4);
    unsigned short* o = out + ((size_t)frag * 32 + lane) * 16;
    const float* w = W + (size_t)n * K + kb;
#pragma unroll
    for (int h = 0; h < 16; ++h) o[h] = f32_to_bf16(w[h]);
}

// ---------------- H init: broadcast emb over batch (f32 + bf16 shadow) -----
__global__ void init_H(const float* __restrict__ emb,
                       float* __restrict__ H,
                       unsigned short* __restrict__ Hb) {
    int t = blockIdx.x * blockDim.x + threadIdx.x;   // over MTOT*D/4
    const int per = (N_ * D_) / 4;
    int pos = t % per;
    float4 v = ((const float4*)emb)[pos];
    ((float4*)H)[t] = v;
    ushort4 o;
    o.x = f32_to_bf16(v.x); o.y = f32_to_bf16(v.y);
    o.z = f32_to_bf16(v.z); o.w = f32_to_bf16(v.w);
    ((ushort4*)Hb)[t] = o;
}

// ---------------- tiled bf16 WMMA GEMM: Out = A @ W^T (+bias, epilogue) ----
// A: [rows][K] bf16 row-major. Bp: fragment-packed weights. 256 thr = 8 waves.
// Block computes 128(M) x 64(N); wave w owns rows mBlock + w*16 .. +15 and all
// 4 N-fragments (4 x v8f accumulators). A panels [128 x 256] staged in LDS
// via async global->LDS copies (no VGPR round-trip).
enum { EPI_GELU_BF16 = 0, EPI_BIAS_BF16 = 1, EPI_BIAS_F32 = 2 };

template <int K, int NOUT, int EPI>
__global__ __launch_bounds__(256)
void gemm_bf16(const unsigned short* __restrict__ A,
               const unsigned short* __restrict__ Bp,
               const float* __restrict__ bias,
               void* __restrict__ Out) {
    constexpr int LDSROW = 264;                       // pad 8 halves / row
    __shared__ unsigned short As[128 * LDSROW];       // ~67.6 KB
    const int tid  = threadIdx.x;
    const int wave = tid >> 5;
    const int lane = tid & 31;
    const int hi   = lane >> 4;                       // 0 | 1
    const int l15  = lane & 15;
    const int mBlock = blockIdx.x * 128;
    const int nBase  = blockIdx.y * 64;

    const v8f vz = {0.f, 0.f, 0.f, 0.f, 0.f, 0.f, 0.f, 0.f};
    v8f acc[4];
    acc[0] = vz; acc[1] = vz; acc[2] = vz; acc[3] = vz;

    for (int kp = 0; kp < K; kp += 256) {
        if (kp) __syncthreads();
        // cooperative, coalesced staging of the 128x256 bf16 A panel into LDS
#pragma unroll
        for (int i = 0; i < 16; ++i) {
            int linear = tid + i * 256;               // 0..4095 16B chunks
            int row = linear >> 5;
            int c16 = linear & 31;
            const unsigned short* g =
                A + (size_t)(mBlock + row) * K + kp + c16 * 8;
#if USE_ASYNC_LDS
            async_ld_b128((unsigned int)(size_t)(&As[row * LDSROW + c16 * 8]), g);
#else
            uint4 v = *(const uint4*)g;
            *(uint4*)(&As[row * LDSROW + c16 * 8]) = v;
#endif
        }
        if (kp + 256 < K)                              // speculative prefetch
            __builtin_prefetch(A + (size_t)(mBlock + (tid >> 1)) * K + kp + 256, 0, 0);
#if USE_ASYNC_LDS
        async_wait_all();                              // s_wait_asynccnt 0
#endif
        __syncthreads();

#pragma unroll
        for (int kt = 0; kt < 8; ++kt) {
            // A fragment, contiguous k (16*hi + s), same order as Bp
            const int arow = wave * 16 + l15;
            const int kb   = kt * 32 + hi * 16;
            v16bf af = load_frag16(&As[arow * LDSROW + kb],
                                   &As[arow * LDSROW + kb + 8]);
#pragma unroll
            for (int nf = 0; nf < 4; ++nf) {
                int fragIdx = (nBase / 16 + nf) * (K / 32) + (kp / 32 + kt);
                const unsigned short* bp = Bp + ((size_t)fragIdx * 32 + lane) * 16;
                v16bf bf = load_frag16(bp, bp + 8);
                acc[nf] = __builtin_amdgcn_wmma_f32_16x16x32_bf16(
                    false, af, false, bf, (short)0, acc[nf], false, false);
            }
        }
    }

    // epilogue: C/D 16x16 f32 layout -> (row = e + hi*8, col = l15)
#pragma unroll
    for (int nf = 0; nf < 4; ++nf) {
        const int col   = nBase + nf * 16 + l15;
        const float bv  = bias[col];
        const int rbase = mBlock + wave * 16 + hi * 8;
#pragma unroll
        for (int e = 0; e < 8; ++e) {
            float v = acc[nf][e] + bv;
            const size_t idx = (size_t)(rbase + e) * NOUT + col;
            if constexpr (EPI == EPI_GELU_BF16) {
                ((unsigned short*)Out)[idx] = f32_to_bf16(gelu_exact(v));
            } else if constexpr (EPI == EPI_BIAS_BF16) {
                ((unsigned short*)Out)[idx] = f32_to_bf16(v);
            } else {
                ((float*)Out)[idx] = v;
            }
        }
    }
}

// ---------------- pointer scatter (deterministic last-dup-wins) ------------
__global__ void winner_init(int* __restrict__ winner) {
    int t = blockIdx.x * blockDim.x + threadIdx.x;    // B*N
    winner[t] = -1;
}

__global__ void scatter_winner(const int* __restrict__ p, int* __restrict__ winner) {
    int t = blockIdx.x * blockDim.x + threadIdx.x;    // B*N
    int b = t >> 12;                                  // N = 4096
    int i = t & (N_ - 1);
    atomicMax(&winner[(b << 12) + p[t]], i);          // last (max i) wins
}

__global__ void gather_agg(const unsigned short* __restrict__ msgs,
                           const int* __restrict__ winner,
                           unsigned short* __restrict__ agg) {
    int t = blockIdx.x * blockDim.x + threadIdx.x;    // B*N*(D/8)
    int c = t & 31;                                   // 8-half chunk in row
    int r = t >> 5;                                   // dest row b*N + j
    int w = winner[r];
    uint4 v = {0u, 0u, 0u, 0u};
    if (w >= 0) {
        int b = r >> 12;
        v = *(const uint4*)(msgs + (size_t)((b << 12) + w) * D_ + c * 8);
    }
    *(uint4*)(agg + (size_t)r * D_ + c * 8) = v;
}

// ---------------- GRU gates (f32), in-place H update + bf16 shadow ---------
__global__ void gru_gate(const float* __restrict__ gi, const float* __restrict__ gh,
                         float* __restrict__ H, unsigned short* __restrict__ Hb,
                         int mOff) {
    int t = blockIdx.x * blockDim.x + threadIdx.x;    // CHUNKM * D
    int row = t >> 8, d = t & (D_ - 1);
    size_t g = (size_t)row * (3 * D_);
    float ir = gi[g + d], iz = gi[g + D_ + d], in_ = gi[g + 2 * D_ + d];
    float hr = gh[g + d], hz = gh[g + D_ + d], hn  = gh[g + 2 * D_ + d];
    size_t hidx = (size_t)(mOff + row) * D_ + d;
    float h = H[hidx];
    float r = 1.f / (1.f + expf(-(ir + hr)));
    float z = 1.f / (1.f + expf(-(iz + hz)));
    float n = tanhf(in_ + r * hn);
    float hnew = (1.f - z) * n + z * h;
    H[hidx]  = hnew;
    Hb[hidx] = f32_to_bf16(hnew);
}

// ---------------- head: combined = GELU([H[s]; emb[k]] @ combW^T + b) ------
__global__ void head_combine(const float* __restrict__ H, const float* __restrict__ emb,
                             const int* __restrict__ s, const int* __restrict__ kq,
                             const float* __restrict__ comb_W, const float* __restrict__ comb_b,
                             float* __restrict__ combined) {
    __shared__ float sq[2 * D_];
    int b = blockIdx.x, d = threadIdx.x;              // 256 threads
    sq[d]      = H[((size_t)b * N_ + s[b]) * D_ + d];
    sq[D_ + d] = emb[(size_t)kq[b] * D_ + d];
    __syncthreads();
    float acc = comb_b[d];
    const float* w = comb_W + (size_t)d * (2 * D_);
#pragma unroll 4
    for (int j = 0; j < 2 * D_; ++j) acc += w[j] * sq[j];
    combined[b * D_ + d] = gelu_exact(acc);
}

__global__ void head_logits(const float* __restrict__ combined,
                            const float* __restrict__ head_W,
                            const float* __restrict__ head_b,
                            float* __restrict__ out) {
    __shared__ float cb[D_];
    int b = blockIdx.y;
    int n = blockIdx.x * 256 + threadIdx.x;
    cb[threadIdx.x] = combined[b * D_ + threadIdx.x];
    __syncthreads();
    float acc = head_b[n];
    const float* w = head_W + (size_t)n * D_;
#pragma unroll 4
    for (int j = 0; j < D_; ++j) acc += cb[j] * w[j];
    out[(size_t)b * N_ + n] = acc;
}

// ---------------- workspace layout (bytes) ---------------------------------
static constexpr size_t SZ_H    = (size_t)MTOT * D_ * 4;          // f32 H
static constexpr size_t SZ_HB   = (size_t)MTOT * D_ * 2;          // bf16 H
static constexpr size_t SZ_HID  = (size_t)CHUNKM * HID_ * 2;      // bf16 hidden chunk
static constexpr size_t SZ_MSG  = (size_t)MTOT * D_ * 2;          // bf16 msgs
static constexpr size_t SZ_AGG  = (size_t)MTOT * D_ * 2;          // bf16 agg
static constexpr size_t SZ_GI   = (size_t)CHUNKM * 3 * D_ * 4;    // f32 gi chunk
static constexpr size_t SZ_GH   = (size_t)CHUNKM * 3 * D_ * 4;    // f32 gh chunk
static constexpr size_t SZ_WIN  = (size_t)B_ * N_ * 4;            // winner idx
static constexpr size_t SZ_CMB  = (size_t)B_ * D_ * 4;            // combined
static constexpr size_t SZ_PW1  = (size_t)HID_ * D_ * 2;
static constexpr size_t SZ_PW2  = (size_t)D_ * HID_ * 2;
static constexpr size_t SZ_PIH  = (size_t)3 * D_ * D_ * 2;
static constexpr size_t SZ_PHH  = (size_t)3 * D_ * D_ * 2;

static constexpr size_t OFF_H   = 0;
static constexpr size_t OFF_HB  = OFF_H   + SZ_H;
static constexpr size_t OFF_HID = OFF_HB  + SZ_HB;
static constexpr size_t OFF_MSG = OFF_HID + SZ_HID;
static constexpr size_t OFF_AGG = OFF_MSG + SZ_MSG;
static constexpr size_t OFF_GI  = OFF_AGG + SZ_AGG;
static constexpr size_t OFF_GH  = OFF_GI  + SZ_GI;
static constexpr size_t OFF_WIN = OFF_GH  + SZ_GH;
static constexpr size_t OFF_CMB = OFF_WIN + SZ_WIN;
static constexpr size_t OFF_PW1 = OFF_CMB + SZ_CMB;
static constexpr size_t OFF_PW2 = OFF_PW1 + SZ_PW1;
static constexpr size_t OFF_PIH = OFF_PW2 + SZ_PW2;
static constexpr size_t OFF_PHH = OFF_PIH + SZ_PIH;

extern "C" void kernel_launch(void* const* d_in, const int* in_sizes, int n_in,
                              void* d_out, int out_size, void* d_ws, size_t ws_size,
                              hipStream_t stream) {
    (void)in_sizes; (void)n_in; (void)out_size; (void)ws_size;

    const int*   p       = (const int*)d_in[0];
    const int*   s       = (const int*)d_in[1];
    const int*   kq      = (const int*)d_in[2];
    const float* emb     = (const float*)d_in[3];
    const float* msg_W1  = (const float*)d_in[4];
    const float* msg_b1  = (const float*)d_in[5];
    const float* msg_W2  = (const float*)d_in[6];
    const float* msg_b2  = (const float*)d_in[7];
    const float* gru_Wih = (const float*)d_in[8];
    const float* gru_Whh = (const float*)d_in[9];
    const float* gru_bih = (const float*)d_in[10];
    const float* gru_bhh = (const float*)d_in[11];
    const float* comb_W  = (const float*)d_in[12];
    const float* comb_b  = (const float*)d_in[13];
    const float* head_W  = (const float*)d_in[14];
    const float* head_b  = (const float*)d_in[15];
    float* out = (float*)d_out;

    char* ws = (char*)d_ws;
    float*          H    = (float*)(ws + OFF_H);
    unsigned short* Hb   = (unsigned short*)(ws + OFF_HB);
    unsigned short* hid  = (unsigned short*)(ws + OFF_HID);
    unsigned short* msgs = (unsigned short*)(ws + OFF_MSG);
    unsigned short* agg  = (unsigned short*)(ws + OFF_AGG);
    float*          gi   = (float*)(ws + OFF_GI);
    float*          gh   = (float*)(ws + OFF_GH);
    int*            win  = (int*)(ws + OFF_WIN);
    float*          cmb  = (float*)(ws + OFF_CMB);
    unsigned short* pW1  = (unsigned short*)(ws + OFF_PW1);
    unsigned short* pW2  = (unsigned short*)(ws + OFF_PW2);
    unsigned short* pIH  = (unsigned short*)(ws + OFF_PIH);
    unsigned short* pHH  = (unsigned short*)(ws + OFF_PHH);

    // 1) repack weights to WMMA fragment layout (bf16)
    repack_weight<<<64, 256, 0, stream>>>(msg_W1,  pW1, HID_,   D_);
    repack_weight<<<64, 256, 0, stream>>>(msg_W2,  pW2, D_,     HID_);
    repack_weight<<<48, 256, 0, stream>>>(gru_Wih, pIH, 3 * D_, D_);
    repack_weight<<<48, 256, 0, stream>>>(gru_Whh, pHH, 3 * D_, D_);

    // 2) H = broadcast(emb)
    init_H<<<(MTOT * D_ / 4) / 256, 256, 0, stream>>>(emb, H, Hb);

    const dim3 gW1(CHUNKM / 128, HID_ / 64);   // (64,16)
    const dim3 gW2(CHUNKM / 128, D_ / 64);     // (64, 4)
    const dim3 gGR(CHUNKM / 128, 3 * D_ / 64); // (64,12)

    for (int layer = 0; layer < L_; ++layer) {
        // message MLP (chunked over rows)
        for (int c = 0; c < NCHUNK; ++c) {
            const size_t off = (size_t)c * CHUNKM * D_;
            gemm_bf16<D_,  HID_, EPI_GELU_BF16><<<gW1, 256, 0, stream>>>(
                Hb + off, pW1, msg_b1, hid);
            gemm_bf16<HID_, D_,  EPI_BIAS_BF16><<<gW2, 256, 0, stream>>>(
                hid, pW2, msg_b2, msgs + off);
        }
        // pointer scatter: agg[b, p[b,i]] = msgs[b,i] (last duplicate wins)
        winner_init<<<(B_ * N_) / 256, 256, 0, stream>>>(win);
        scatter_winner<<<(B_ * N_) / 256, 256, 0, stream>>>(p, win);
        gather_agg<<<(B_ * N_ * (D_ / 8)) / 256, 256, 0, stream>>>(msgs, win, agg);
        // GRU (chunked over rows)
        for (int c = 0; c < NCHUNK; ++c) {
            const size_t off = (size_t)c * CHUNKM * D_;
            gemm_bf16<D_, 3 * D_, EPI_BIAS_F32><<<gGR, 256, 0, stream>>>(
                agg + off, pIH, gru_bih, gi);
            gemm_bf16<D_, 3 * D_, EPI_BIAS_F32><<<gGR, 256, 0, stream>>>(
                Hb + off, pHH, gru_bhh, gh);
            gru_gate<<<(CHUNKM * D_) / 256, 256, 0, stream>>>(
                gi, gh, H, Hb, c * CHUNKM);
        }
    }

    // head
    head_combine<<<B_, 256, 0, stream>>>(H, emb, s, kq, comb_W, comb_b, cmb);
    head_logits<<<dim3(N_ / 256, B_), 256, 0, stream>>>(cmb, head_W, head_b, out);
}
